// ContinuousSpatialSSM_V4_13357348291152
// MI455X (gfx1250) — matrix operand order repaired and checked by reading, not compiled, not run on hardware
//
#include <hip/hip_runtime.h>
#include <cstdint>
#include <cstddef>

// ---------------------------------------------------------------------------
// ContinuousSpatialSSM_V4 for MI455X (gfx1250, wave32, WMMA, 320KB LDS/WGP)
//   B=4, N=1024 (32x32), D=384, S=16, K_steps=2
// ---------------------------------------------------------------------------

#define B_SZ     4
#define N_TOK    1024
#define D_INNER  384
#define S_STATE  16
#define M_ROWS   (B_SZ * N_TOK)      // 4096
#define GRID_W   32                  // sqrt(N_TOK)
#define DT_MAXF  0.15f
#define LD_PROJ  48                  // [B_mat | C_real | C_imag] fused columns

typedef __attribute__((ext_vector_type(16))) _Float16 v16h;
typedef __attribute__((ext_vector_type(8)))  _Float16 v8h;
typedef __attribute__((ext_vector_type(8)))  float    v8f;

__device__ __forceinline__ float softplusf(float v) {
  return v > 20.f ? v : log1pf(expf(v));
}

// ---------------------------------------------------------------------------
// Kernel 1: convert f32 inputs to f16 operand buffers for WMMA.
//   xh     : x as f16, row-major (M_ROWS, 384)
//   wselfh : dt_self_W as f16, row-major (n, k)  (used as B^T: out = x @ W.T)
//   wdiffh : dt_diff_W as f16, row-major (n, k)
//   wproj  : [B_W^T ; C_real_W^T ; C_imag_W^T] as f16, (48, 384) row-major
// ---------------------------------------------------------------------------
__global__ void k_pack_f16(const float* __restrict__ x,
                           const float* __restrict__ wself,
                           const float* __restrict__ wdiff,
                           const float* __restrict__ BW,
                           const float* __restrict__ CrW,
                           const float* __restrict__ CiW,
                           _Float16* __restrict__ xh,
                           _Float16* __restrict__ wselfh,
                           _Float16* __restrict__ wdiffh,
                           _Float16* __restrict__ wproj)
{
  const int NX = M_ROWS * D_INNER;        // 1,572,864
  const int NW = D_INNER * D_INNER;       //   147,456
  const int NP = LD_PROJ * D_INNER;       //    18,432
  for (int i = blockIdx.x * blockDim.x + threadIdx.x; i < NX;
       i += gridDim.x * blockDim.x) {
    xh[i] = (_Float16)x[i];
    if (i < NW) {
      wselfh[i] = (_Float16)wself[i];
      wdiffh[i] = (_Float16)wdiff[i];
    }
    if (i < NP) {
      int j = i / D_INNER;                // output feature 0..47
      int k = i - j * D_INNER;            // reduction index
      float v;
      if (j < 16)      v = BW [k * S_STATE + j];
      else if (j < 32) v = CrW[k * S_STATE + (j - 16)];
      else             v = CiW[k * S_STATE + (j - 32)];
      wproj[i] = (_Float16)v;
    }
  }
}

// ---------------------------------------------------------------------------
// Kernel 2: WMMA GEMM tile kernel, N-blocked.
//   out[m, n] = act( sum_k xh[m,k] * wh[n,k] + bias[n] )
//   - each wave computes a 16 x (16*NBLK) output strip: A tile loaded once per
//     k-step, reused for NBLK v_wmma_f32_16x16x32_f16 ops (VGPR-resident acc)
//   - wh is (n, k) row-major so the B operand is 16 contiguous halves per lane
//   - act: softplus + min(DT_MAX) for the delta projections
// ---------------------------------------------------------------------------
template <int NBLK>
__global__ __launch_bounds__(128)
void k_gemm_wmma(const _Float16* __restrict__ xh,
                 const _Float16* __restrict__ wh,
                 const float*    __restrict__ bias,
                 float*          __restrict__ out,
                 int ldOut, int act)
{
  const int K     = D_INNER;
  const int wave  = threadIdx.x >> 5;
  const int lane  = threadIdx.x & 31;
  const int mt    = blockIdx.x * 4 + wave;     // 0..255 (grid.x = 64)
  const int n0    = blockIdx.y * 16 * NBLK;
  const int m0    = mt * 16;
  const int mloc  = lane & 15;
  const int khalf = lane >> 4;                 // 0 or 1

  v8f acc[NBLK];
#pragma unroll
  for (int t = 0; t < NBLK; ++t) acc[t] = (v8f){};

  for (int kk = 0; kk < K; kk += 32) {
    // A operand: 16x32 f16.  lane(khalf,m): 16 contiguous K at kk + khalf*16
    const _Float16* ap = xh + (size_t)(m0 + mloc) * K + kk + khalf * 16;
    v8h alo = *(const v8h*)ap;
    v8h ahi = *(const v8h*)(ap + 8);
    v16h A;
#pragma unroll
    for (int i = 0; i < 8; ++i) { A[i] = alo[i]; A[i + 8] = ahi[i]; }

#pragma unroll
    for (int t = 0; t < NBLK; ++t) {
      // B operand: 32x16 f16. lane(khalf,n): 16 contiguous K at kk + khalf*16
      const _Float16* bp =
          wh + (size_t)(n0 + t * 16 + mloc) * K + kk + khalf * 16;
      v16h Bv = *(const v16h*)bp;
      acc[t] = __builtin_amdgcn_wmma_f32_16x16x32_f16(
          /*neg_a=*/false, A, /*neg_b=*/false, Bv,
          /*c_mod=*/(short)0, acc[t], /*reuse_a=*/false, /*reuse_b=*/false);
    }
  }

  // D layout: VGPR r -> row m0 + r + khalf*8, col n0 + t*16 + mloc
#pragma unroll
  for (int t = 0; t < NBLK; ++t) {
    const int col = n0 + t * 16 + mloc;
#pragma unroll
    for (int r = 0; r < 8; ++r) {
      const int row = m0 + r + khalf * 8;
      float v = acc[t][r];
      if (bias) v += bias[col];
      if (act)  v = fminf(softplusf(v), DT_MAXF);
      out[(size_t)row * ldOut + col] = v;
    }
  }
}

// ---------------------------------------------------------------------------
// Kernel 3: the full 2-step reaction-diffusion recurrence + output contraction,
// one workgroup per (d, b).  The entire (N=1024, S=16) complex state slab for
// this (b,d) lives in 128 KB of LDS (fits the 320 KB/WGP budget) so the
// Laplacian neighbor exchange and both K-steps never touch HBM.
//   LDS layout [s][n]: consecutive threads -> consecutive banks (no conflicts,
//   including the n+-1 / n+-32 neighbor reads).
//   proj = fused (B,N,48) buffer: [B_mat | C_real | C_imag]
// ---------------------------------------------------------------------------
__global__ __launch_bounds__(1024)
void k_pde(const float* __restrict__ x,
           const float* __restrict__ delta_s,
           const float* __restrict__ delta_d,
           const float* __restrict__ proj,
           const float* __restrict__ A_log,
           const float* __restrict__ conv_w,
           const float* __restrict__ D_param,
           const float* __restrict__ r_alpha,
           const float* __restrict__ r_beta,
           float* __restrict__ y)
{
  extern __shared__ float lds[];
  float* hr = lds;                       // [16][1024]
  float* hi = lds + S_STATE * N_TOK;     // [16][1024]

  const int n  = threadIdx.x;            // spatial position 0..1023
  const int d  = blockIdx.x;             // 0..383
  const int b  = blockIdx.y;             // 0..3
  const int bn = b * N_TOK + n;
  const float* prow = proj + (size_t)bn * LD_PROJ;

  // C_real/C_imag are consumed only after the recurrence: warm L2/L1 now
  // (lowers to global_prefetch_b8; non-blocking, no counter use).
  __builtin_prefetch(prow + 16, 0, 3);

  const float xv = x      [(size_t)bn * D_INNER + d];
  const float ds = delta_s[(size_t)bn * D_INNER + d];
  const float dd = delta_d[(size_t)bn * D_INNER + d];

  float mi[S_STATE], Aa[S_STATE];
#pragma unroll
  for (int s = 0; s < S_STATE; ++s) {
    const float bm = prow[s];            // B_mat
    Aa[s] = -softplusf(A_log[d * S_STATE + s]);
    mi[s] = xv * bm;                     // mamba_input == initial h_real
    hr[s * N_TOK + n] = mi[s];
    hi[s * N_TOK + n] = 0.f;
  }

  float w[9];
#pragma unroll
  for (int t = 0; t < 9; ++t) w[t] = conv_w[d * 9 + t];
  const float al_d = r_alpha[d];
  const float be_d = r_beta[d];

  // neighbor indices with replicate ("edge") padding on the 32x32 grid
  const int r  = n >> 5, c = n & 31;
  const int rm = (r > 0)  ? r - 1 : 0;
  const int rp = (r < 31) ? r + 1 : 31;
  const int cm = (c > 0)  ? c - 1 : 0;
  const int cp = (c < 31) ? c + 1 : 31;
  const int i00 = rm * GRID_W + cm, i01 = rm * GRID_W + c, i02 = rm * GRID_W + cp;
  const int i10 = r  * GRID_W + cm,                        i12 = r  * GRID_W + cp;
  const int i20 = rp * GRID_W + cm, i21 = rp * GRID_W + c, i22 = rp * GRID_W + cp;

  __syncthreads();

  const float dt = 0.5f;                 // 1 / K_steps
  for (int step = 0; step < 2; ++step) {
#pragma unroll 1
    for (int s = 0; s < S_STATE; ++s) {  // each s-plane is independent
      const float* Hr = hr + s * N_TOK;
      const float* Hi = hi + s * N_TOK;
      const float hrv = Hr[n];
      const float hiv = Hi[n];
      const float lr = w[0]*Hr[i00] + w[1]*Hr[i01] + w[2]*Hr[i02]
                     + w[3]*Hr[i10] + w[4]*hrv     + w[5]*Hr[i12]
                     + w[6]*Hr[i20] + w[7]*Hr[i21] + w[8]*Hr[i22];
      const float li = w[0]*Hi[i00] + w[1]*Hi[i01] + w[2]*Hi[i02]
                     + w[3]*Hi[i10] + w[4]*hiv     + w[5]*Hi[i12]
                     + w[6]*Hi[i20] + w[7]*Hi[i21] + w[8]*Hi[i22];
      const float f1r = ds * (Aa[s] * hrv + mi[s]);
      const float f1i = ds * (Aa[s] * hiv);
      const float f2r = -dd * li;
      const float f2i =  dd * lr;
      const float rs  = al_d - be_d * (hrv * hrv + hiv * hiv);
      const float nrv = hrv + dt * (f1r + f2r + hrv * rs);
      const float niv = hiv + dt * (f1i + f2i + hiv * rs);
      __syncthreads();                   // everyone done reading plane s
      hr[s * N_TOK + n] = nrv;
      hi[s * N_TOK + n] = niv;
      __syncthreads();                   // writes visible before next use
    }
  }

  // y[b,n,d] = sum_s hr*C_real + hi*C_imag + x*D
  float acc = xv * D_param[d];
#pragma unroll
  for (int s = 0; s < S_STATE; ++s) {
    acc += hr[s * N_TOK + n] * prow[16 + s]     // C_real
         + hi[s * N_TOK + n] * prow[32 + s];    // C_imag
  }
  y[(size_t)bn * D_INNER + d] = acc;
}

// ---------------------------------------------------------------------------
// Host-side orchestration
// ---------------------------------------------------------------------------
extern "C" void kernel_launch(void* const* d_in, const int* in_sizes, int n_in,
                              void* d_out, int out_size, void* d_ws, size_t ws_size,
                              hipStream_t stream) {
  (void)in_sizes; (void)n_in; (void)out_size; (void)ws_size;

  const float* x        = (const float*)d_in[0];
  const float* dt_selfW = (const float*)d_in[1];
  const float* dt_selfb = (const float*)d_in[2];
  const float* dt_diffW = (const float*)d_in[3];
  const float* dt_diffb = (const float*)d_in[4];
  const float* B_W      = (const float*)d_in[5];
  const float* Cr_W     = (const float*)d_in[6];
  const float* Ci_W     = (const float*)d_in[7];
  const float* D_param  = (const float*)d_in[8];
  const float* A_log    = (const float*)d_in[9];
  const float* conv_w   = (const float*)d_in[10];
  const float* r_alpha  = (const float*)d_in[11];
  const float* r_beta   = (const float*)d_in[12];
  // d_in[13] = K_steps (known constant 2)
  float* y = (float*)d_out;

  // workspace carve-up (256B aligned slabs), total ~16.5 MB
  char* ws = (char*)d_ws;
  size_t off = 0;
  auto carve = [&](size_t bytes) -> void* {
    void* p = ws + off;
    off += (bytes + 255) & ~(size_t)255;
    return p;
  };
  _Float16* xh      = (_Float16*)carve((size_t)M_ROWS * D_INNER * 2);
  _Float16* wselfh  = (_Float16*)carve((size_t)D_INNER * D_INNER * 2);
  _Float16* wdiffh  = (_Float16*)carve((size_t)D_INNER * D_INNER * 2);
  _Float16* wproj   = (_Float16*)carve((size_t)LD_PROJ * D_INNER * 2);
  float*    delta_s = (float*)   carve((size_t)M_ROWS * D_INNER * 4);
  float*    delta_d = (float*)   carve((size_t)M_ROWS * D_INNER * 4);
  float*    proj    = (float*)   carve((size_t)M_ROWS * LD_PROJ * 4);

  // 1) f32 -> f16 operand packing
  k_pack_f16<<<2048, 256, 0, stream>>>(x, dt_selfW, dt_diffW, B_W, Cr_W, Ci_W,
                                       xh, wselfh, wdiffh, wproj);

  // 2) delta projections: (4096x384) @ (384x384)^T, softplus + clamp.
  //    N-blocked x4: each wave computes 16x64, A reused for 4 WMMAs.
  {
    dim3 grid(64, 6);   // 256 M-tiles x 6 N-strips (64 cols each)
    k_gemm_wmma<4><<<grid, 128, 0, stream>>>(xh, wselfh, dt_selfb, delta_s,
                                             D_INNER, /*act=*/1);
    k_gemm_wmma<4><<<grid, 128, 0, stream>>>(xh, wdiffh, dt_diffb, delta_d,
                                             D_INNER, /*act=*/1);
  }

  // 3) fused skinny projections: (4096x384) @ (384x48) -> [B_mat|C_real|C_imag]
  //    one launch, A reused for 3 WMMAs per k-step.
  {
    dim3 grid(64, 1);
    k_gemm_wmma<3><<<grid, 128, 0, stream>>>(xh, wproj, nullptr, proj,
                                             LD_PROJ, /*act=*/0);
  }

  // 4) two-step recurrence + output contraction, 128 KB LDS per workgroup
  {
    (void)hipFuncSetAttribute(reinterpret_cast<const void*>(k_pde),
                              hipFuncAttributeMaxDynamicSharedMemorySize,
                              2 * S_STATE * N_TOK * (int)sizeof(float));
    dim3 grid(D_INNER, B_SZ);  // (384, 4)
    size_t lds_bytes = (size_t)2 * S_STATE * N_TOK * sizeof(float); // 131072
    k_pde<<<grid, N_TOK, lds_bytes, stream>>>(x, delta_s, delta_d, proj,
                                              A_log, conv_w, D_param,
                                              r_alpha, r_beta, y);
  }
}
